// Decoder_43344809951905
// MI455X (gfx1250) — matrix operand order repaired and checked by reading, not compiled, not run on hardware
//
#include <hip/hip_runtime.h>
#include <hip/hip_bf16.h>
#include <math.h>

typedef _Float16 f16x16 __attribute__((ext_vector_type(16)));
typedef _Float16 f16x8  __attribute__((ext_vector_type(8)));
typedef float    f32x8  __attribute__((ext_vector_type(8)));

#define BB   64
#define TT   1024
#define DD2  512
#define HHD  256
#define GG   1024
#define LLB  96
#define TBR  (BB * TT)     /* 65536 rows */
#define XK   640           /* 612 padded to 640 */

// ---------------------------------------------------------------------------
// WMMA helpers (16x16x32 f16 -> f32), CDNA5 wave32 layouts
// A fragment (16x32, row-major source): lane L row = L%16;
//   elems 0..7  -> K = 8*(L/16) + e       (contiguous)
//   elems 8..15 -> K = 16 + 8*(L/16) + e  (contiguous)
// B fragment (32x16): lane L col N = L%16; elem e -> same K as A elem e (8*(L/16)+e; 16+8*(L/16)+e-8 for e>=8)
// C/D (16x16 f32): lane L col N = L%16; vgpr v -> row M = v + 8*(L/16)
// ---------------------------------------------------------------------------

__device__ __forceinline__ f32x8 wmma_f16(f16x16 a, f16x16 b, f32x8 c) {
  return __builtin_amdgcn_wmma_f32_16x16x32_f16(false, a, false, b, (short)0, c,
                                                false, false);
}

__device__ __forceinline__ f16x16 afrag_f16(const _Float16* base, int row0,
                                            int ldk, int kbase, int lane) {
  const _Float16* p = base + (size_t)(row0 + (lane & 15)) * ldk + kbase +
                      ((lane & 16) >> 1);
  f16x8 lo = *(const f16x8*)p;
  f16x8 hi = *(const f16x8*)(p + 16);
  f16x16 a;
#pragma unroll
  for (int e = 0; e < 8; ++e) { a[e] = lo[e]; a[8 + e] = hi[e]; }
  return a;
}

__device__ __forceinline__ f16x16 afrag_f32cvt(const float* p) {
  f32x8 lo = *(const f32x8*)p;
  f32x8 hi = *(const f32x8*)(p + 16);
  f16x16 a;
#pragma unroll
  for (int e = 0; e < 8; ++e) {
    a[e]     = (_Float16)lo[e];
    a[8 + e] = (_Float16)hi[e];
  }
  return a;
}

__device__ __forceinline__ f16x16 bfrag(const _Float16* wb, int nt, int ksteps,
                                        int ks, int lane) {
  return *(const f16x16*)(wb + (((size_t)(nt * ksteps + ks) * 32 + lane) << 4));
}

__device__ __forceinline__ float sigmoidf_(float x) {
  return 1.0f / (1.0f + __expf(-x));
}

// ---------------------------------------------------------------------------
// K0: swizzle a row-major f32 weight W[N][Kld] (out[n] = sum_k in[k]*W[n][k])
// into f16 B-fragment layout [ntile][kstep][lane][16]. k >= Kreal -> 0.
// ---------------------------------------------------------------------------
__global__ void swizzle_w_k(const float* __restrict__ W, _Float16* __restrict__ dst,
                            int ntiles, int ksteps, int ld, int Kreal) {
  int id = blockIdx.x * blockDim.x + threadIdx.x;
  if (id >= ntiles * ksteps * 32) return;
  int lane = id & 31;
  int ks = (id >> 5) % ksteps;
  int n = ((id >> 5) / ksteps) * 16 + (lane & 15);
  int kb = ks * 32 + ((lane & 16) ? 8 : 0);
  _Float16* o = dst + ((size_t)id << 4);
#pragma unroll
  for (int e = 0; e < 16; ++e) {
    int k = kb + ((e < 8) ? e : 8 + e);
    o[e] = (_Float16)((k < Kreal) ? W[(size_t)n * ld + k] : 0.0f);
  }
}

__global__ void zero_bar_k(unsigned* bar) { *bar = 0u; }

// ---------------------------------------------------------------------------
// K1: prefix sums P[b][t][d] (P[b][0][d]=0, P[b][t+1][d]=sum enc[b][:t+1][d])
// ---------------------------------------------------------------------------
__global__ void prefix_k(const float* __restrict__ enc, float* __restrict__ P) {
  int id = blockIdx.x * blockDim.x + threadIdx.x;  // 64*512
  int b = id >> 9, d = id & 511;
  float s = 0.0f;
  P[((size_t)b * (TT + 1)) * DD2 + d] = 0.0f;
  const float* e = enc + ((size_t)b * TT) * DD2 + d;
  float* p = P + ((size_t)b * (TT + 1) + 1) * DD2 + d;
  for (int t = 0; t < TT; ++t) {
    s += e[(size_t)t * DD2];
    p[(size_t)t * DD2] = s;
  }
}

// ---------------------------------------------------------------------------
// K2: build X f16 [r = t*64+b][640] = [pe(100) | avg(512) | pad(28)]
// ---------------------------------------------------------------------------
__global__ void buildx_k(const float* __restrict__ P,
                         const float* __restrict__ pos_table,
                         const int* __restrict__ pos_id,
                         const int* __restrict__ start_index,
                         _Float16* __restrict__ X) {
  int r = blockIdx.x;            // 65536
  int t = r >> 6, b = r & 63;
  int pp = (t == 0) ? 0 : pos_id[(t - 1) * BB + b];
  int ps = (t == 0) ? 0 : start_index[(t - 1) * BB + b];
  bool valid = (ps >= 0) && (t > 0);
  int start = min(ps, t - 1);
  start = max(start, 0);
  start = min(start, TT - 1);
  float invlen = 1.0f / (float)max(t - start, 1);
  const float* Pt = P + ((size_t)b * (TT + 1) + t) * DD2;
  const float* Ps = P + ((size_t)b * (TT + 1) + start) * DD2;
  _Float16* xr = X + (size_t)r * XK;
  for (int c = threadIdx.x; c < XK; c += blockDim.x) {
    float val;
    if (c < 100) val = pos_table[pp * 100 + c];
    else if (c < 612) {
      int d = c - 100;
      val = valid ? (Pt[d] - Ps[d]) * invlen : 0.0f;
    } else val = 0.0f;
    xr[c] = (_Float16)val;
  }
}

// ---------------------------------------------------------------------------
// K3: Z[r][256] = tanh(X @ comb_W^T + comb_b); rows with t==0 -> 0
// Register-blocked 1Mx4N; all fragment loads of a k-step hoisted ahead of the
// WMMA burst; 2x k-unroll so next step's loads overlap current WMMAs.
// ---------------------------------------------------------------------------
__global__ void gemm_z_k(const _Float16* __restrict__ X,
                         const _Float16* __restrict__ combB,
                         const float* __restrict__ comb_b,
                         _Float16* __restrict__ Z) {
  int wave = (blockIdx.x * blockDim.x + threadIdx.x) >> 5;  // 16384
  int lane = threadIdx.x & 31;
  int mt = wave >> 2;       // 4096
  int ntg = wave & 3;       // 4 groups of 4 N-tiles
  f32x8 acc[4] = {};
#pragma unroll 2
  for (int ks = 0; ks < 20; ++ks) {
    f16x16 a = afrag_f16(X, mt * 16, XK, ks * 32, lane);
    f16x16 bf[4];
#pragma unroll
    for (int j = 0; j < 4; ++j) bf[j] = bfrag(combB, ntg * 4 + j, 20, ks, lane);
#pragma unroll
    for (int j = 0; j < 4; ++j) acc[j] = wmma_f16(a, bf[j], acc[j]);
  }
  int rbase = mt * 16 + ((lane & 16) ? 8 : 0);
  bool zrow = (mt < 4);  // rows 0..63 are t==0
#pragma unroll
  for (int j = 0; j < 4; ++j) {
    int n = (ntg * 4 + j) * 16 + (lane & 15);
    float bias = comb_b[n];
#pragma unroll
    for (int v = 0; v < 8; ++v) {
      float zv = zrow ? 0.0f : tanhf(acc[j][v] + bias);
      Z[(size_t)(rbase + v) * HHD + n] = (_Float16)zv;
    }
  }
}

// ---------------------------------------------------------------------------
// K4: gates_pre[r][1024] = Z @ W_ih^T + b_ih + b_hh   (f32)
// ---------------------------------------------------------------------------
__global__ void gemm_gates_k(const _Float16* __restrict__ Z,
                             const _Float16* __restrict__ ihB,
                             const float* __restrict__ b_ih,
                             const float* __restrict__ b_hh,
                             float* __restrict__ gates_pre) {
  int wave = (blockIdx.x * blockDim.x + threadIdx.x) >> 5;  // 65536
  int lane = threadIdx.x & 31;
  int mt = wave >> 4;       // 4096
  int ntg = wave & 15;      // 16 groups of 4 N-tiles
  f32x8 acc[4] = {};
#pragma unroll 2
  for (int ks = 0; ks < 8; ++ks) {
    f16x16 a = afrag_f16(Z, mt * 16, HHD, ks * 32, lane);
    f16x16 bf[4];
#pragma unroll
    for (int j = 0; j < 4; ++j) bf[j] = bfrag(ihB, ntg * 4 + j, 8, ks, lane);
#pragma unroll
    for (int j = 0; j < 4; ++j) acc[j] = wmma_f16(a, bf[j], acc[j]);
  }
  int rbase = mt * 16 + ((lane & 16) ? 8 : 0);
#pragma unroll
  for (int j = 0; j < 4; ++j) {
    int n = (ntg * 4 + j) * 16 + (lane & 15);
    float bias = b_ih[n] + b_hh[n];
#pragma unroll
    for (int v = 0; v < 8; ++v)
      gates_pre[(size_t)(rbase + v) * GG + n] = acc[j][v] + bias;
  }
}

// ---------------------------------------------------------------------------
// K5: recurrence. 16 blocks x 128 threads = 64 waves. Block owns hidden tile
// jt = blockIdx; wave owns batch tile m. Block's W_hh B-fragments (4 gates x
// 8 ksteps, 32KB) staged once in LDS; c-state lives in WMMA accumulator
// registers for all 1024 steps. Per step: gates = gates_pre[t] + h@W_hh^T
// (k-loop fully unrolled so all loads front-run the WMMA chain), LSTM
// elementwise, h_t -> h_all[t], prefetch next gates_pre slab, grid barrier.
// ---------------------------------------------------------------------------
__global__ void __launch_bounds__(128) recurrence_k(
    const float* __restrict__ gates_pre, const _Float16* __restrict__ hhB,
    _Float16* __restrict__ h_all, unsigned* bar) {
  __shared__ _Float16 lbs[4 * 8 * 32 * 16];  // 32 KB: [q][ks][lane][16]
  int lane = threadIdx.x & 31;
  int m = threadIdx.x >> 5;   // batch tile 0..3
  int jt = blockIdx.x;        // hidden tile 0..15
  int n0 = jt * 16 + (lane & 15);
  int rloc = m * 16 + ((lane & 16) ? 8 : 0);

  // stage this block's W_hh fragments into LDS (each wave: 8 fragments)
#pragma unroll
  for (int k8 = 0; k8 < 8; ++k8) {
    int f = m * 8 + k8;           // 0..31
    int q = f >> 3, ks = f & 7;
    const _Float16* src =
        hhB + (((size_t)((q * 16 + jt) * 8 + ks) * 32 + lane) << 4);
    *(f16x16*)(lbs + (((q * 8 + ks) * 32 + lane) << 4)) = *(const f16x16*)src;
  }
  __syncthreads();

  f32x8 c = {};
  for (int t = 0; t < TT; ++t) {
    f32x8 g4[4];
    const float* gp = gates_pre + (size_t)t * BB * GG;
#pragma unroll
    for (int q = 0; q < 4; ++q)
#pragma unroll
      for (int v = 0; v < 8; ++v)
        g4[q][v] = gp[(size_t)(rloc + v) * GG + q * HHD + n0];
    // prefetch next step's gates_pre slab (256KB; one 128B line per thread)
    if (t + 1 < TT) {
      int gtid = blockIdx.x * 128 + threadIdx.x;  // 0..2047
      __builtin_prefetch(gates_pre + (size_t)(t + 1) * BB * GG + gtid * 32, 0, 1);
    }
    if (t > 0) {
      const _Float16* hprev = h_all + (size_t)(t - 1) * BB * HHD;
#pragma unroll
      for (int ks = 0; ks < 8; ++ks) {
        f16x16 a = afrag_f16(hprev, m * 16, HHD, ks * 32, lane);
        f16x16 bq[4];
#pragma unroll
        for (int q = 0; q < 4; ++q)
          bq[q] = *(const f16x16*)(lbs + (((q * 8 + ks) * 32 + lane) << 4));
#pragma unroll
        for (int q = 0; q < 4; ++q) g4[q] = wmma_f16(a, bq[q], g4[q]);
      }
    }
    _Float16* hcur = h_all + (size_t)t * BB * HHD;
#pragma unroll
    for (int v = 0; v < 8; ++v) {
      float ig = sigmoidf_(g4[0][v]);
      float fg = sigmoidf_(g4[1][v]);
      float gg = tanhf(g4[2][v]);
      float og = sigmoidf_(g4[3][v]);
      c[v] = fg * c[v] + ig * gg;
      float hv = og * tanhf(c[v]);
      hcur[(size_t)(rloc + v) * HHD + n0] = (_Float16)hv;
    }
    // grid barrier across 16 blocks (monotonic counter, device scope)
    __syncthreads();
    if (threadIdx.x == 0) {
      __threadfence();
      __hip_atomic_fetch_add(bar, 1u, __ATOMIC_RELEASE, __HIP_MEMORY_SCOPE_AGENT);
      unsigned target = 16u * (unsigned)(t + 1);
      while (__hip_atomic_load(bar, __ATOMIC_ACQUIRE, __HIP_MEMORY_SCOPE_AGENT) <
             target) {
        __builtin_amdgcn_s_sleep(1);
      }
      __threadfence();
    }
    __syncthreads();
  }
}

// ---------------------------------------------------------------------------
// K6: out[b*T+t][96] = [h_t | enc[b][t]] @ lin_W^T    (f32 scores)
// Register-blocked: one wave computes all 6 label tiles for its M-tile.
// ---------------------------------------------------------------------------
__global__ void gemm_out_k(const _Float16* __restrict__ h_all,
                           const float* __restrict__ enc,
                           const _Float16* __restrict__ linB,
                           float* __restrict__ out) {
  int mt = (blockIdx.x * blockDim.x + threadIdx.x) >> 5;  // 4096
  int lane = threadIdx.x & 31;
  f32x8 acc[6] = {};
#pragma unroll 2
  for (int ks = 0; ks < 24; ++ks) {
    f16x16 a;
    if (ks < 8) {
      a = afrag_f16(h_all, mt * 16, HHD, ks * 32, lane);
    } else {
      int r = mt * 16 + (lane & 15);  // r = t*64+b
      const float* p = enc + ((size_t)(r & 63) * TT + (r >> 6)) * DD2 +
                       (ks - 8) * 32 + ((lane & 16) >> 1);
      a = afrag_f32cvt(p);
    }
    f16x16 bf[6];
#pragma unroll
    for (int j = 0; j < 6; ++j) bf[j] = bfrag(linB, j, 24, ks, lane);
#pragma unroll
    for (int j = 0; j < 6; ++j) acc[j] = wmma_f16(a, bf[j], acc[j]);
  }
  int rbase = mt * 16 + ((lane & 16) ? 8 : 0);
#pragma unroll
  for (int j = 0; j < 6; ++j) {
    int l = j * 16 + (lane & 15);
#pragma unroll
    for (int v = 0; v < 8; ++v) {
      int r = rbase + v;                      // r = t*64+b
      int b_ = r & 63, t_ = r >> 6;
      out[((size_t)b_ * TT + t_) * LLB + l] = acc[j][v];
    }
  }
}

// ---------------------------------------------------------------------------
// K7: in-place masked log_softmax over 96 labels. One wave per row.
// row = b*T+t; mask: (t==0 && l==APP_ID(1)) -> -1e10 before softmax.
// ---------------------------------------------------------------------------
__global__ void logsoftmax_k(float* __restrict__ out) {
  int row = (blockIdx.x << 3) | (threadIdx.x >> 5);
  int lane = threadIdx.x & 31;
  float* p = out + (size_t)row * LLB;
  int t = row & (TT - 1);
  float v[3];
#pragma unroll
  for (int i = 0; i < 3; ++i) {
    int l = i * 32 + lane;
    float x = p[l];
    if (t == 0 && l == 1) x = -1e10f;
    v[i] = x;
  }
  float mx = fmaxf(fmaxf(v[0], v[1]), v[2]);
#pragma unroll
  for (int off = 16; off >= 1; off >>= 1) mx = fmaxf(mx, __shfl_xor(mx, off));
  float s = __expf(v[0] - mx) + __expf(v[1] - mx) + __expf(v[2] - mx);
#pragma unroll
  for (int off = 16; off >= 1; off >>= 1) s += __shfl_xor(s, off);
  float lse = mx + logf(s);
#pragma unroll
  for (int i = 0; i < 3; ++i) p[i * 32 + lane] = v[i] - lse;
}

// ---------------------------------------------------------------------------
static inline size_t al256(size_t x) { return (x + 255) & ~(size_t)255; }

extern "C" void kernel_launch(void* const* d_in, const int* in_sizes, int n_in,
                              void* d_out, int out_size, void* d_ws,
                              size_t ws_size, hipStream_t stream) {
  const float* enc        = (const float*)d_in[0];
  const float* pos_table  = (const float*)d_in[1];
  const float* W_ih       = (const float*)d_in[2];
  const float* b_ih       = (const float*)d_in[3];
  const float* W_hh       = (const float*)d_in[4];
  const float* b_hh       = (const float*)d_in[5];
  const float* comb_W     = (const float*)d_in[6];
  const float* comb_b     = (const float*)d_in[7];
  const float* lin_W      = (const float*)d_in[8];
  const int*   pos_id     = (const int*)d_in[9];
  const int*   start_idx  = (const int*)d_in[10];
  float* out = (float*)d_out;
  char* ws = (char*)d_ws;

  size_t offP     = 0;                                           // 134,348,800 B
  size_t offGates = al256(offP + (size_t)BB * (TT + 1) * DD2 * 4);
  size_t offX     = offGates;  // X (f16, 84MB) aliases gates_pre (268MB) region
  size_t offZ     = al256(offGates + (size_t)TBR * GG * 4);
  size_t offH     = al256(offZ + (size_t)TBR * HHD * 2);
  size_t offCombB = al256(offH + (size_t)TBR * HHD * 2);
  size_t offIhB   = al256(offCombB + (size_t)16 * 20 * 32 * 16 * 2);
  size_t offHhB   = al256(offIhB + (size_t)64 * 8 * 32 * 16 * 2);
  size_t offLinB  = al256(offHhB + (size_t)64 * 8 * 32 * 16 * 2);
  size_t offBar   = al256(offLinB + (size_t)6 * 24 * 32 * 16 * 2);

  float*     P      = (float*)(ws + offP);
  float*     gatesP = (float*)(ws + offGates);
  _Float16*  X      = (_Float16*)(ws + offX);
  _Float16*  Z      = (_Float16*)(ws + offZ);
  _Float16*  h_all  = (_Float16*)(ws + offH);
  _Float16*  combB  = (_Float16*)(ws + offCombB);
  _Float16*  ihB    = (_Float16*)(ws + offIhB);
  _Float16*  hhB    = (_Float16*)(ws + offHhB);
  _Float16*  linB   = (_Float16*)(ws + offLinB);
  unsigned*  bar    = (unsigned*)(ws + offBar);

  // weight swizzles (B-fragment layout)
  swizzle_w_k<<<(16 * 20 * 32 + 255) / 256, 256, 0, stream>>>(comb_W, combB, 16, 20, 612, 612);
  swizzle_w_k<<<(64 * 8 * 32 + 255) / 256, 256, 0, stream>>>(W_ih, ihB, 64, 8, 256, 256);
  swizzle_w_k<<<(64 * 8 * 32 + 255) / 256, 256, 0, stream>>>(W_hh, hhB, 64, 8, 256, 256);
  swizzle_w_k<<<(6 * 24 * 32 + 255) / 256, 256, 0, stream>>>(lin_W, linB, 6, 24, 768, 768);
  zero_bar_k<<<1, 1, 0, stream>>>(bar);

  // parallel pre-pass
  prefix_k<<<(BB * DD2) / 256, 256, 0, stream>>>(enc, P);
  buildx_k<<<TBR, 128, 0, stream>>>(P, pos_table, pos_id, start_idx, X);
  gemm_z_k<<<2048, 256, 0, stream>>>(X, combB, comb_b, Z);        // 16384 waves
  gemm_gates_k<<<8192, 256, 0, stream>>>(Z, ihB, b_ih, b_hh, gatesP);  // 65536 waves

  // sequential recurrence (persistent, grid-synced, W_hh in LDS)
  recurrence_k<<<16, 128, 0, stream>>>(gatesP, hhB, h_all, bar);

  // parallel post-pass
  gemm_out_k<<<512, 256, 0, stream>>>(h_all, enc, linB, out);     // 4096 waves
  logsoftmax_k<<<TBR / 8, 256, 0, stream>>>(out);
}
